// MultiBondRetention_88132728914058
// MI455X (gfx1250) — matrix operand order repaired and checked by reading, not compiled, not run on hardware
//
#include <hip/hip_runtime.h>
#include <hip/hip_bf16.h>

// ---------------- problem constants (from reference) ----------------
#define M_SEG   128
#define S_LEN   512
#define HD      512
#define NHEAD   8
#define DHEAD   64
#define NTOK    (M_SEG * S_LEN)          // 65536
#define NGROUP  8
#define CGRP    (HD / NGROUP)            // 64
#define GAMMA_F 0.96875f

// ---------------- WMMA types / helpers ----------------
typedef __attribute__((ext_vector_type(16))) __bf16 v16bf;
typedef __attribute__((ext_vector_type(8)))  __bf16 v8bf;
typedef __attribute__((ext_vector_type(8)))  float  v8f;
typedef __attribute__((ext_vector_type(4)))  unsigned int u32x4;
typedef __attribute__((ext_vector_type(8)))  int i32x8;
typedef __attribute__((ext_vector_type(4)))  int i32x4;

static __device__ __forceinline__ __bf16 f2bf(float f) { return (__bf16)f; }

static __device__ __forceinline__ v8f zero_v8f() {
  v8f z;
#pragma unroll
  for (int i = 0; i < 8; ++i) z[i] = 0.f;
  return z;
}

// generic pointer into LDS -> byte offset (LDS aperture keeps offset in addr[31:0])
static __device__ __forceinline__ unsigned ldsoff(const void* p) {
  return (unsigned)(unsigned long long)p;
}

// ---- Tensor Data Mover: DMA a 2D tile (tile_w bf16 elems x tile_h rows, row stride
// row_stride elems in global) into LDS at lds_off, with optional per-row LDS padding.
// pad_icode: pad_interval code (3 = every 16 dwords, 4 = every 32 dwords)
// pad_code:  pad_amount  code (3 = 4 dwords = 16 bytes)     [CDNA5 D# tables]
static __device__ __forceinline__ void tdm_load_2d(unsigned lds_off, const void* gptr,
                                                   int tile_w, int tile_h,
                                                   int row_stride, int pad_icode,
                                                   int pad_code) {
  unsigned long long ga = (unsigned long long)gptr;
  const unsigned TD_HI = 0x4000u;  // tensor_dim0/1 = 1<<30 (no OOB clipping needed)
  u32x4 g0;
  g0[0] = 1u;                                              // count=1, user descriptor
  g0[1] = lds_off;                                         // lds_addr
  g0[2] = (unsigned)(ga & 0xffffffffu);                    // global_addr[31:0]
  g0[3] = (unsigned)((ga >> 32) & 0x01ffffffu) | (2u << 30);  // addr[56:32] | type=2
  i32x8 g1;
  g1[0] = (int)((1u << 16) |                               // data_size = 2 bytes
                (1u << 20) |                               // pad_enable
                ((unsigned)pad_icode << 22) | ((unsigned)pad_code << 25));
  g1[1] = 0;                                               // barrier addr | dim0 lo16 (0)
  g1[2] = (int)TD_HI;                                      // dim0 hi16 | dim1 lo16 (0)
  g1[3] = (int)(TD_HI | ((unsigned)tile_w << 16));         // dim1 hi16 | tile_dim0
  g1[4] = tile_h;                                          // tile_dim1 | tile_dim2=0
  g1[5] = row_stride;                                      // tensor_dim0_stride[31:0]
  g1[6] = 0;
  g1[7] = 0;
  i32x4 z4;
  z4[0] = z4[1] = z4[2] = z4[3] = 0;
#if defined(__clang_major__) && (__clang_major__ >= 23)
  i32x8 z8;
#pragma unroll
  for (int i = 0; i < 8; ++i) z8[i] = 0;
  __builtin_amdgcn_tensor_load_to_lds(g0, g1, z4, z4, z8, 0);
#else
  __builtin_amdgcn_tensor_load_to_lds(g0, g1, z4, z4, 0);
#endif
}

// A-matrix fragment 16x32 bf16, row-major source (ld elements, ld % 8 == 0).
static __device__ __forceinline__ v16bf load_a16x32(const __bf16* src, int ld, int lane) {
  const int m = lane & 15, hi = lane >> 4;
  const __bf16* p = src + m * ld + hi * 8;
  v8bf lo = *(const v8bf*)p;
  v8bf hh = *(const v8bf*)(p + 16);
  v16bf a;
#pragma unroll
  for (int j = 0; j < 8; ++j) { a[j] = lo[j]; a[j + 8] = hh[j]; }
  return a;
}

// B-matrix fragment 32x16 bf16 from TRANSPOSED storage Bt[n][k] (row-major, ld % 8 == 0).
static __device__ __forceinline__ v16bf load_bt16x32(const __bf16* srcT, int ld, int lane) {
  const int n = lane & 15, hi = lane >> 4;
  const __bf16* p = srcT + n * ld + hi * 8;
  v8bf lo = *(const v8bf*)p;
  v8bf hh = *(const v8bf*)(p + 16);
  v16bf b;
#pragma unroll
  for (int j = 0; j < 8; ++j) { b[j] = lo[j]; b[j + 8] = hh[j]; }
  return b;
}

static __device__ __forceinline__ v8f wmma_bf16(v16bf a, v16bf b, v8f c) {
  return __builtin_amdgcn_wmma_f32_16x16x32_bf16(false, a, false, b, (short)0, c,
                                                 false, false);
}

// ---------------- kernel 0a: fp32 weights -> bf16 (Wq,Wk,Wv,Wo packed) ---------------
__global__ void wconv_kernel(const float* __restrict__ w0, const float* __restrict__ w1,
                             const float* __restrict__ w2, const float* __restrict__ w3,
                             __bf16* __restrict__ dst) {
  const float* srcs[4] = {w0, w1, w2, w3};
  const float* s = srcs[blockIdx.y];
  __bf16* d = dst + (size_t)blockIdx.y * (HD * HD);
  const int c8 = blockIdx.x * 256 + threadIdx.x;
  const float4* sp = (const float4*)s + (size_t)c8 * 2;
  float4 a = sp[0], b = sp[1];
  v8bf o;
  o[0] = f2bf(a.x); o[1] = f2bf(a.y); o[2] = f2bf(a.z); o[3] = f2bf(a.w);
  o[4] = f2bf(b.x); o[5] = f2bf(b.y); o[6] = f2bf(b.z); o[7] = f2bf(b.w);
  *((v8bf*)d + c8) = o;
}

// ---------------- kernel 0b: message fp32 -> bf16 once ----------------
__global__ void xconv_kernel(const float* __restrict__ x, __bf16* __restrict__ xb) {
  const size_t c8 = (size_t)blockIdx.x * 256 + threadIdx.x;
  const float4* sp = (const float4*)x + c8 * 2;
  float4 a = sp[0], b = sp[1];
  v8bf o;
  o[0] = f2bf(a.x); o[1] = f2bf(a.y); o[2] = f2bf(a.z); o[3] = f2bf(a.w);
  o[4] = f2bf(b.x); o[5] = f2bf(b.y); o[6] = f2bf(b.z); o[7] = f2bf(b.w);
  *((v8bf*)xb + c8) = o;
}

// ---------------- kernel 1: q/k/v = x @ W^T, TDM double-buffered tiles ---------------
// grid: (NTOK/128, HD/64, 3)  block: 256 (8 waves); each wave: 16 rows x 64 cols
__global__ void qkv_kernel(const __bf16* __restrict__ xb, const __bf16* __restrict__ wb,
                           __bf16* __restrict__ outb) {
  __shared__ alignas(16) __bf16 Xs[2][128][40];
  __shared__ alignas(16) __bf16 Ws[2][64][40];   // Ws[o][c] == Bt for B = W^T
  const int tid = threadIdx.x, lane = tid & 31, w = tid >> 5;
  const int row0 = blockIdx.x * 128, col0 = blockIdx.y * 64;
  const __bf16* wsel = wb + (size_t)blockIdx.z * (HD * HD);
  __bf16* osel = outb + (size_t)blockIdx.z * ((size_t)NTOK * HD);
  const bool issuer = (tid < 32);                // wave 0 drives the TDM

  v8f acc[4];
#pragma unroll
  for (int t = 0; t < 4; ++t) acc[t] = zero_v8f();

  if (issuer) {                                  // prefetch first tiles
    tdm_load_2d(ldsoff(&Xs[0][0][0]), &xb[(size_t)row0 * HD], 32, 128, HD, 3, 3);
    tdm_load_2d(ldsoff(&Ws[0][0][0]), &wsel[(size_t)col0 * HD], 32, 64, HD, 3, 3);
  }
  int buf = 0;
  for (int kk = 0; kk < HD; kk += 32, buf ^= 1) {
    if (issuer) __builtin_amdgcn_s_wait_tensorcnt(0);
    __syncthreads();                             // tile `buf` visible to all waves
    if (issuer && kk + 32 < HD) {                // DMA next tile during compute
      tdm_load_2d(ldsoff(&Xs[buf ^ 1][0][0]), &xb[(size_t)row0 * HD + kk + 32],
                  32, 128, HD, 3, 3);
      tdm_load_2d(ldsoff(&Ws[buf ^ 1][0][0]), &wsel[(size_t)col0 * HD + kk + 32],
                  32, 64, HD, 3, 3);
    }
    v16bf a = load_a16x32(&Xs[buf][w * 16][0], 40, lane);
#pragma unroll
    for (int nt = 0; nt < 4; ++nt) {
      v16bf b = load_bt16x32(&Ws[buf][nt * 16][0], 40, lane);
      acc[nt] = wmma_bf16(a, b, acc[nt]);
    }
  }
  const int hi = lane >> 4, n = lane & 15;
#pragma unroll
  for (int nt = 0; nt < 4; ++nt)
#pragma unroll
    for (int r = 0; r < 8; ++r)
      osel[(size_t)(row0 + w * 16 + r + 8 * hi) * HD + col0 + nt * 16 + n] =
          f2bf(acc[nt][r]);
}

// ---------------- kernel 2: A[m,h] = K^T V  (per-segment state contribution) ---------
__global__ void kv_state_kernel(const __bf16* __restrict__ kb,
                                const __bf16* __restrict__ vb,
                                float* __restrict__ Abuf) {
  __shared__ alignas(16) __bf16 Kt[64][40];   // Kt[d1][s] : A = K^T
  __shared__ alignas(16) __bf16 Vt[64][40];   // Vt[d2][s] : Bt for B = V
  const int tid = threadIdx.x, lane = tid & 31, w = tid >> 5;
  const int m = blockIdx.x >> 3, h = blockIdx.x & 7;
  const size_t segbase = (size_t)m * S_LEN * HD + (size_t)h * DHEAD;

  v8f acc[4];
#pragma unroll
  for (int t = 0; t < 4; ++t) acc[t] = zero_v8f();

  for (int s0 = 0; s0 < S_LEN; s0 += 32) {
    __syncthreads();
#pragma unroll
    for (int i = 0; i < 2; ++i) {                // [32 s][64 d] tile: 256 chunks of 8
      int chunk = tid + i * 128, el = chunk * 8;
      int ss = el >> 6, dd = el & 63;
      size_t g = segbase + (size_t)(s0 + ss) * HD + dd;
      v8bf k8 = *(const v8bf*)&kb[g];
      v8bf v8_ = *(const v8bf*)&vb[g];
#pragma unroll
      for (int j = 0; j < 8; ++j) { Kt[dd + j][ss] = k8[j]; Vt[dd + j][ss] = v8_[j]; }
    }
    __syncthreads();
    v16bf a = load_a16x32(&Kt[w * 16][0], 40, lane);
#pragma unroll
    for (int nt = 0; nt < 4; ++nt) {
      v16bf b = load_bt16x32(&Vt[nt * 16][0], 40, lane);
      acc[nt] = wmma_bf16(a, b, acc[nt]);
    }
  }
  float* out = Abuf + (size_t)blockIdx.x * (DHEAD * DHEAD);
  const int hi = lane >> 4, n = lane & 15;
#pragma unroll
  for (int nt = 0; nt < 4; ++nt)
#pragma unroll
    for (int r = 0; r < 8; ++r)
      out[(w * 16 + r + 8 * hi) * DHEAD + nt * 16 + n] = acc[nt][r];
}

// ---------------- kernel 3: elementwise scan over segments (in place) ----------------
__global__ void scan_kernel(float* __restrict__ Abuf) {
  const int e = blockIdx.x * 256 + threadIdx.x;          // 0..32767
  const float decay = powf(GAMMA_F, (float)S_LEN);
  float r = 0.f;
  for (int m = 0; m < M_SEG; ++m) {
    float a = Abuf[(size_t)m * (NHEAD * DHEAD * DHEAD) + e];
    r = a + decay * r;
    Abuf[(size_t)m * (NHEAD * DHEAD * DHEAD) + e] = r;
  }
}

// ---------------- kernel 4: retention per (segment, head, 64-query tile) -------------
// h = relu( (Q K^T * scale) V + gamma^(m+1) * (Q r[m]) ), streamed over key tiles
// grid: (S/64, NHEAD, M_SEG), 128 threads (4 waves, 16 query rows each)
__global__ void attn_kernel(const __bf16* __restrict__ qb, const __bf16* __restrict__ kb,
                            const __bf16* __restrict__ vb, const float* __restrict__ Abuf,
                            __bf16* __restrict__ hb) {
  __shared__ alignas(16) __bf16 Qs[64][72];        // [q][d]
  __shared__ alignas(16) __bf16 Ks[64][72];        // [t][d]  == Bt for B = K^T
  __shared__ alignas(16) __bf16 Vt[64][72];        // [d][t]  == Bt for B = V
  __shared__ alignas(16) __bf16 Rt[64][72];        // [d2][d1]== Bt for B = r
  __shared__ alignas(16) __bf16 Ps[4][16][72];     // per-wave score staging [q][t]
  const int tid = threadIdx.x, lane = tid & 31, w = tid >> 5;
  const int q0 = blockIdx.x * 64, h = blockIdx.y, m = blockIdx.z;
  const size_t segbase = (size_t)m * S_LEN * HD + (size_t)h * DHEAD;
  const float scale = 0.125f;                      // d^-0.5
  const bool issuer = (tid < 32);

  // TDM the Q tile; meanwhile everyone stages r[m,h] transposed as bf16
  if (issuer)
    tdm_load_2d(ldsoff(&Qs[0][0]), &qb[segbase + (size_t)q0 * HD], 64, 64, HD, 4, 3);
  const float* rsrc = Abuf + ((size_t)m * NHEAD + h) * (DHEAD * DHEAD);
#pragma unroll
  for (int i = 0; i < 8; ++i) {                    // 1024 chunks of 4 fp32
    int chunk = tid + i * 128, el = chunk * 4;
    int d1 = el >> 6, d2 = el & 63;
    float4 rv = *(const float4*)&rsrc[d1 * DHEAD + d2];
    Rt[d2 + 0][d1] = f2bf(rv.x); Rt[d2 + 1][d1] = f2bf(rv.y);
    Rt[d2 + 2][d1] = f2bf(rv.z); Rt[d2 + 3][d1] = f2bf(rv.w);
  }
  if (issuer) __builtin_amdgcn_s_wait_tensorcnt(0);
  __syncthreads();

  v8f acc_i[4], acc_c[4];
#pragma unroll
  for (int t = 0; t < 4; ++t) { acc_i[t] = zero_v8f(); acc_c[t] = zero_v8f(); }

  // cross = Q @ r
#pragma unroll
  for (int kt = 0; kt < 2; ++kt) {
    v16bf a = load_a16x32(&Qs[w * 16][kt * 32], 72, lane);
#pragma unroll
    for (int nt = 0; nt < 4; ++nt) {
      v16bf b = load_bt16x32(&Rt[nt * 16][kt * 32], 72, lane);
      acc_c[nt] = wmma_bf16(a, b, acc_c[nt]);
    }
  }

  // intra = (Q K^T * scale) @ V, streamed over 64-wide key tiles
  for (int t0 = 0; t0 < S_LEN; t0 += 64) {
    __syncthreads();                               // previous tile fully consumed
    if (issuer)                                    // K tile via TDM ...
      tdm_load_2d(ldsoff(&Ks[0][0]), &kb[segbase + (size_t)t0 * HD], 64, 64, HD, 4, 3);
#pragma unroll
    for (int i = 0; i < 4; ++i) {                  // ... V tile transposed by the waves
      int chunk = tid + i * 128, el = chunk * 8;
      int r = el >> 6, c = el & 63;
      v8bf vv = *(const v8bf*)&vb[segbase + (size_t)(t0 + r) * HD + c];
#pragma unroll
      for (int j = 0; j < 8; ++j) Vt[c + j][r] = vv[j];
    }
    if (issuer) __builtin_amdgcn_s_wait_tensorcnt(0);
    __syncthreads();

    v8f p[4];
#pragma unroll
    for (int t = 0; t < 4; ++t) p[t] = zero_v8f();
#pragma unroll
    for (int kt = 0; kt < 2; ++kt) {
      v16bf a = load_a16x32(&Qs[w * 16][kt * 32], 72, lane);
#pragma unroll
      for (int nt = 0; nt < 4; ++nt) {
        v16bf b = load_bt16x32(&Ks[nt * 16][kt * 32], 72, lane);
        p[nt] = wmma_bf16(a, b, p[nt]);
      }
    }
    // relayout C-frag -> A-frag through per-wave LDS tile (no cross-wave dependency)
    {
      const int hi = lane >> 4, n = lane & 15;
#pragma unroll
      for (int nt = 0; nt < 4; ++nt)
#pragma unroll
        for (int r = 0; r < 8; ++r)
          Ps[w][r + 8 * hi][nt * 16 + n] = f2bf(p[nt][r] * scale);
    }
#pragma unroll
    for (int kt = 0; kt < 2; ++kt) {
      v16bf a = load_a16x32(&Ps[w][0][kt * 32], 72, lane);
#pragma unroll
      for (int nt = 0; nt < 4; ++nt) {
        v16bf b = load_bt16x32(&Vt[nt * 16][kt * 32], 72, lane);
        acc_i[nt] = wmma_bf16(a, b, acc_i[nt]);
      }
    }
  }

  const float g = powf(GAMMA_F, (float)(m + 1));
  const int hi = lane >> 4, n = lane & 15;
#pragma unroll
  for (int nt = 0; nt < 4; ++nt)
#pragma unroll
    for (int r = 0; r < 8; ++r) {
      float hv = acc_i[nt][r] + g * acc_c[nt][r];
      hv = hv > 0.f ? hv : 0.f;                                  // ReLU
      hb[segbase + (size_t)(q0 + w * 16 + r + 8 * hi) * HD + nt * 16 + n] = f2bf(hv);
    }
}

// ---------------- kernel 5: res = x + h @ Wo^T + bo -> d_out rows 1..N ---------------
__global__ void outproj_kernel(const __bf16* __restrict__ hb, const __bf16* __restrict__ wob,
                               const float* __restrict__ x, const float* __restrict__ bo,
                               float* __restrict__ out) {
  __shared__ alignas(16) __bf16 Hs[2][128][40];
  __shared__ alignas(16) __bf16 Ws[2][64][40];
  const int tid = threadIdx.x, lane = tid & 31, w = tid >> 5;
  const int row0 = blockIdx.x * 128, col0 = blockIdx.y * 64;
  const bool issuer = (tid < 32);

  v8f acc[4];
#pragma unroll
  for (int t = 0; t < 4; ++t) acc[t] = zero_v8f();

  if (issuer) {
    tdm_load_2d(ldsoff(&Hs[0][0][0]), &hb[(size_t)row0 * HD], 32, 128, HD, 3, 3);
    tdm_load_2d(ldsoff(&Ws[0][0][0]), &wob[(size_t)col0 * HD], 32, 64, HD, 3, 3);
  }
  int buf = 0;
  for (int kk = 0; kk < HD; kk += 32, buf ^= 1) {
    if (issuer) __builtin_amdgcn_s_wait_tensorcnt(0);
    __syncthreads();
    if (issuer && kk + 32 < HD) {
      tdm_load_2d(ldsoff(&Hs[buf ^ 1][0][0]), &hb[(size_t)row0 * HD + kk + 32],
                  32, 128, HD, 3, 3);
      tdm_load_2d(ldsoff(&Ws[buf ^ 1][0][0]), &wob[(size_t)col0 * HD + kk + 32],
                  32, 64, HD, 3, 3);
    }
    v16bf a = load_a16x32(&Hs[buf][w * 16][0], 40, lane);
#pragma unroll
    for (int nt = 0; nt < 4; ++nt) {
      v16bf b = load_bt16x32(&Ws[buf][nt * 16][0], 40, lane);
      acc[nt] = wmma_bf16(a, b, acc[nt]);
    }
  }
  const int hi = lane >> 4, n = lane & 15;
#pragma unroll
  for (int nt = 0; nt < 4; ++nt)
#pragma unroll
    for (int r = 0; r < 8; ++r) {
      int row = row0 + w * 16 + r + 8 * hi, col = col0 + nt * 16 + n;
      float res = acc[nt][r] + bo[col] + x[(size_t)row * HD + col];
      out[(size_t)(row + 1) * HD + col] = res;   // +1: zero row prepended
    }
}

// ---------------- kernel 6: in-place GroupNorm per (segment, group) + zero row 0 -----
__global__ void gnorm_kernel(float* __restrict__ out, const float* __restrict__ gw,
                             const float* __restrict__ gb) {
  __shared__ float rs[256], rs2[256];
  const int tid = threadIdx.x;
  const int m = blockIdx.x >> 3, g = blockIdx.x & 7;
  float* base = out + (size_t)(1 + m * S_LEN) * HD + g * CGRP;
  float s = 0.f, s2 = 0.f;
  for (int i = tid; i < S_LEN * CGRP; i += 256) {
    int r = i >> 6, c = i & 63;
    float v = base[(size_t)r * HD + c];
    s += v; s2 += v * v;
  }
  rs[tid] = s; rs2[tid] = s2;
  __syncthreads();
  for (int st = 128; st > 0; st >>= 1) {
    if (tid < st) { rs[tid] += rs[tid + st]; rs2[tid] += rs2[tid + st]; }
    __syncthreads();
  }
  const float inv_n = 1.f / (float)(S_LEN * CGRP);
  const float mean = rs[0] * inv_n;
  const float var = rs2[0] * inv_n - mean * mean;
  const float inv = rsqrtf(var + 1e-5f);
  for (int i = tid; i < S_LEN * CGRP; i += 256) {
    int r = i >> 6, c = i & 63;
    float v = base[(size_t)r * HD + c];
    base[(size_t)r * HD + c] = (v - mean) * inv * gw[g * CGRP + c] + gb[g * CGRP + c];
  }
  if (blockIdx.x == 0)
    for (int i = tid; i < HD; i += 256) out[i] = 0.f;   // cached zero vector row
}

// ---------------- host launcher ----------------
extern "C" void kernel_launch(void* const* d_in, const int* in_sizes, int n_in,
                              void* d_out, int out_size, void* d_ws, size_t ws_size,
                              hipStream_t stream) {
  (void)in_sizes; (void)n_in; (void)out_size; (void)ws_size;
  const float* msg = (const float*)d_in[0];
  const float* Wq  = (const float*)d_in[1];
  const float* Wk  = (const float*)d_in[2];
  const float* Wv  = (const float*)d_in[3];
  const float* Wo  = (const float*)d_in[4];
  const float* bo  = (const float*)d_in[5];
  const float* gw  = (const float*)d_in[6];
  const float* gb  = (const float*)d_in[7];
  float* out = (float*)d_out;

  char* ws = (char*)d_ws;
  const size_t qkv_bytes = (size_t)NTOK * HD * sizeof(unsigned short); // 64 MiB each
  __bf16* qb = (__bf16*)(ws + 0 * qkv_bytes);
  __bf16* kb = (__bf16*)(ws + 1 * qkv_bytes);
  __bf16* vb = (__bf16*)(ws + 2 * qkv_bytes);
  __bf16* hb = (__bf16*)(ws + 3 * qkv_bytes);
  __bf16* xb = (__bf16*)(ws + 4 * qkv_bytes);
  float*  Abuf = (float*)(ws + 5 * qkv_bytes);                         // 16 MiB
  __bf16* wqkv = (__bf16*)(ws + 5 * qkv_bytes +
                           (size_t)M_SEG * NHEAD * DHEAD * DHEAD * sizeof(float));
  __bf16* wob = wqkv + 3 * (size_t)HD * HD;

  wconv_kernel<<<dim3(HD * HD / (256 * 8), 4), 256, 0, stream>>>(Wq, Wk, Wv, Wo, wqkv);
  xconv_kernel<<<dim3((size_t)NTOK * HD / (256 * 8)), 256, 0, stream>>>(msg, xb);
  qkv_kernel<<<dim3(NTOK / 128, HD / 64, 3), 256, 0, stream>>>(xb, wqkv, qb);
  kv_state_kernel<<<dim3(M_SEG * NHEAD), 128, 0, stream>>>(kb, vb, Abuf);
  scan_kernel<<<dim3((NHEAD * DHEAD * DHEAD) / 256), 256, 0, stream>>>(Abuf);
  attn_kernel<<<dim3(S_LEN / 64, NHEAD, M_SEG), 128, 0, stream>>>(qb, kb, vb, Abuf, hb);
  outproj_kernel<<<dim3(NTOK / 128, HD / 64), 256, 0, stream>>>(hb, wob, msg, bo, out);
  gnorm_kernel<<<dim3(M_SEG * NGROUP), 256, 0, stream>>>(out, gw, gb);
}
